// GNNAttention_32272384262237
// MI455X (gfx1250) — matrix operand order repaired and checked by reading, not compiled, not run on hardware
//
#include <hip/hip_runtime.h>
#include <math.h>

typedef _Float16 f16;
typedef __attribute__((ext_vector_type(16))) _Float16 v16h;
typedef __attribute__((ext_vector_type(8)))  _Float16 v8h;
typedef __attribute__((ext_vector_type(8)))  float    v8f;

#define NPTS 2048
#define NB   4
#define DIM  32
#define KNN  16
#define NN   (NB * NPTS)   // 8192 nodes per graph
#define HID  64
#define EPSV 1e-5f

// ---------------- WMMA helpers (CDNA5 wave32, 16x16x32 f16) ----------------

__device__ inline v8f wmma16(v16h a, v16h b, v8f c) {
  return __builtin_amdgcn_wmma_f32_16x16x32_f16(false, a, false, b, (short)0, c,
                                                false, false);
}

// A tile 16x32 (MxK) from row-major f16, leading dim ld.
// lane<16: row=lane, K={0..15}; lane>=16: row=lane-16, K={16..31}
// -> two contiguous 16B loads per lane.
__device__ inline v16h load_a_f16(const f16* base, int ld, int lane) {
  int row = lane & 15;
  int kb  = (lane & 16) ? 16 : 0;
  const f16* p = base + row * ld + kb;
  v8h lo = *(const v8h*)(p);
  v8h hi = *(const v8h*)(p + 8);
  return __builtin_shufflevector(lo, hi, 0, 1, 2, 3, 4, 5, 6, 7, 8, 9, 10, 11,
                                 12, 13, 14, 15);
}

// Same pattern from row-major f32 (two 32B loads + convert).
__device__ inline v16h load_a_f32(const float* base, int ld, int lane) {
  int row = lane & 15;
  int kb  = (lane & 16) ? 16 : 0;
  const float* p = base + row * ld + kb;
  v8h lo = __builtin_convertvector(*(const v8f*)(p), v8h);
  v8h hi = __builtin_convertvector(*(const v8f*)(p + 8), v8h);
  return __builtin_shufflevector(lo, hi, 0, 1, 2, 3, 4, 5, 6, 7, 8, 9, 10, 11,
                                 12, 13, 14, 15);
}

// B fragment from pre-swizzled weights: one contiguous 32B load per lane.
// Swizzled layout: swz[((tile*32 + lane)<<4) + e] = W[kt*32 + (lane&16) + e][nt*16 + (lane&15)]
__device__ inline v16h load_b_swz(const f16* sW, int tile, int lane) {
  return *(const v16h*)(sW + (((tile << 5) + lane) << 4));
}

// B = srcT tile (B[k][n] = src[n*ld + k]): row-contiguous -> one 32B load.
__device__ inline v16h load_bT_row(const f16* base, int ld, int lane) {
  int n  = lane & 15;
  int kh = lane & 16;
  return *(const v16h*)(base + n * ld + kh);
}

// ---------------- kNN graph (per-point argmin x16 via LDS) ----------------

__global__ __launch_bounds__(256) void knn_kernel(const float* __restrict__ xyz,
                                                  int* __restrict__ nbr) {
  __shared__ float sd2[NPTS];
  __shared__ float smin[256];
  __shared__ int   sidx[256];
  int p = blockIdx.x;          // global point id
  int b = p >> 11, i = p & (NPTS - 1);
  const float* base = xyz + (size_t)b * NPTS * 3;
  float xi = base[i * 3 + 0], yi = base[i * 3 + 1], zi = base[i * 3 + 2];
  for (int j = threadIdx.x; j < NPTS; j += 256) {
    float dx = base[j * 3 + 0] - xi;
    float dy = base[j * 3 + 1] - yi;
    float dz = base[j * 3 + 2] - zi;
    sd2[j] = dx * dx + dy * dy + dz * dz;
  }
  __syncthreads();
  if (threadIdx.x == 0) sd2[i] = 1e30f;  // exclude self
  __syncthreads();
  for (int t = 0; t < KNN; ++t) {
    float best = 1e30f; int bi = 0;
    for (int j = threadIdx.x; j < NPTS; j += 256)
      if (sd2[j] < best) { best = sd2[j]; bi = j; }
    smin[threadIdx.x] = best; sidx[threadIdx.x] = bi;
    __syncthreads();
    for (int s = 128; s > 0; s >>= 1) {
      if (threadIdx.x < s && smin[threadIdx.x + s] < smin[threadIdx.x]) {
        smin[threadIdx.x] = smin[threadIdx.x + s];
        sidx[threadIdx.x] = sidx[threadIdx.x + s];
      }
      __syncthreads();
    }
    if (threadIdx.x == 0) {
      int c = sidx[0];
      nbr[p * KNN + t] = b * NPTS + c;   // global flat neighbor id
      sd2[c] = 1e30f;
    }
    __syncthreads();
  }
}

// ---------------- CGConv: gated messages, max aggregation ----------------

__global__ __launch_bounds__(256) void cgconv_kernel(
    const float* __restrict__ x, const int* __restrict__ nbr,
    const float* __restrict__ Wf, const float* __restrict__ bfv,
    const float* __restrict__ Ws, const float* __restrict__ bsv,
    float* __restrict__ agg) {
  int gid  = blockIdx.x * blockDim.x + threadIdx.x;
  int node = gid >> 5;
  int lane = threadIdx.x & 31;
  if (node >= NN) return;
  float xi = x[node * DIM + lane];
  float bfl = bfv[lane], bsl = bsv[lane];
  float out = -INFINITY;
  for (int e = 0; e < KNN; ++e) {
    int j = nbr[node * KNN + e];
    float xj = x[j * DIM + lane];
    float af = bfl, as = bsl;
#pragma unroll
    for (int c = 0; c < 32; ++c) {          // z[0:32] = x_i (dst)
      float zc = __shfl(xi, c, 32);
      af += zc * Wf[c * DIM + lane];
      as += zc * Ws[c * DIM + lane];
    }
#pragma unroll
    for (int c = 0; c < 32; ++c) {          // z[32:64] = x_j (src)
      float zc = __shfl(xj, c, 32);
      af += zc * Wf[(32 + c) * DIM + lane];
      as += zc * Ws[(32 + c) * DIM + lane];
    }
    float sig = 1.0f / (1.0f + __expf(-af));
    float sp  = fmaxf(as, 0.0f) + log1pf(__expf(-fabsf(as)));
    out = fmaxf(out, sig * sp);
  }
  agg[node * DIM + lane] = isfinite(out) ? out : 0.0f;
}

__global__ __launch_bounds__(256) void bn_stats_kernel(const float* __restrict__ agg,
                                                       float* __restrict__ stats) {
  __shared__ float S[256], SS[256];
  int tid = threadIdx.x, ch = tid & 31, grp = tid >> 5;
  float s = 0.f, ss = 0.f;
  for (int r = grp; r < NN; r += 8) {
    float v = agg[r * DIM + ch];
    s += v; ss += v * v;
  }
  S[tid] = s; SS[tid] = ss;
  __syncthreads();
  if (grp == 0) {
    for (int g = 1; g < 8; ++g) { s += S[g * 32 + ch]; ss += SS[g * 32 + ch]; }
    float inv = 1.0f / (float)NN;
    float mu  = s * inv;
    stats[ch]      = mu;
    stats[32 + ch] = ss * inv - mu * mu;
  }
}

__global__ __launch_bounds__(256) void bn_apply_kernel(
    const float* __restrict__ x, const float* __restrict__ agg,
    const float* __restrict__ stats, const float* __restrict__ gamma,
    const float* __restrict__ beta, float* __restrict__ h) {
  int i = blockIdx.x * blockDim.x + threadIdx.x;
  if (i >= NN * DIM) return;
  int ch = i & 31;
  float mu = stats[ch], var = stats[32 + ch];
  h[i] = x[i] + (agg[i] - mu) * rsqrtf(var + EPSV) * gamma[ch] + beta[ch];
}

// f32 -> f16 copy, plus transposed f16 copy hT[b][d][n] = h[b][n][d]
__global__ __launch_bounds__(256) void tof16_tr_kernel(const float* __restrict__ s,
                                                       f16* __restrict__ d,
                                                       f16* __restrict__ dT) {
  int i = blockIdx.x * blockDim.x + threadIdx.x;
  if (i >= NN * DIM) return;
  f16 v = (f16)s[i];
  d[i] = v;
  int ch = i & 31, node = i >> 5;
  int b = node >> 11, n = node & (NPTS - 1);
  dT[((size_t)b * DIM + ch) * NPTS + n] = v;
}

// Pre-swizzle weights (layers*Kdim*Ndim f32) into fragment-major f16 layout.
__global__ __launch_bounds__(256) void swz_b_kernel(const float* __restrict__ W,
                                                    f16* __restrict__ out,
                                                    int Kdim, int Ndim, int total) {
  int i = blockIdx.x * blockDim.x + threadIdx.x;
  if (i >= total) return;
  int per = Kdim * Ndim;
  int l = i / per, j = i - l * per;
  int e = j & 15, lane = (j >> 4) & 31, tile = j >> 9;
  int NT = Ndim >> 4;
  int kt = tile / NT, nt = tile - kt * NT;
  int k = kt * 32 + (lane & 16) + e;
  int n = nt * 16 + (lane & 15);
  out[i] = (f16)W[(size_t)l * per + k * Ndim + n];
}

// ---------------- Edge MLP + segment_sum (wave = 1 node, 16 edges) ----------------
// msg[i] = sum_e ( relu([h_src||h_dst] W1 + b1) W2 + b2 )

__global__ __launch_bounds__(256) void edge_mlp_kernel(
    const f16* __restrict__ hh, const int* __restrict__ nbr,
    const f16* __restrict__ W1s, const float* __restrict__ b1,
    const f16* __restrict__ W2s, const float* __restrict__ b2,
    float* __restrict__ msg) {
  __shared__ f16  sW1[HID * HID];       // swizzled 64x64
  __shared__ f16  sW2[HID * HID];       // swizzled 64x64
  __shared__ f16  sA[8][16 * HID];      // per-wave gathered edge features
  __shared__ float sM[8][16 * HID];     // per-wave hidden activations
  int tid = threadIdx.x;
  for (int i = tid; i < HID * HID / 8; i += 256) {
    ((v8h*)sW1)[i] = ((const v8h*)W1s)[i];
    ((v8h*)sW2)[i] = ((const v8h*)W2s)[i];
  }
  int w = tid >> 5, lane = tid & 31;
  int node = blockIdx.x * 8 + w;
  // stage A as 8-element vector chunks: 16 rows x 8 chunks
  for (int idx = lane; idx < 128; idx += 32) {
    int r = idx >> 3, cc = (idx & 7) << 3;
    v8h v = (cc < 32)
        ? *(const v8h*)(hh + (size_t)nbr[node * KNN + r] * DIM + cc)   // h[src]
        : *(const v8h*)(hh + (size_t)node * DIM + (cc - 32));          // h[dst]
    *(v8h*)(&sA[w][r * HID + cc]) = v;
  }
  __syncthreads();
  int n = lane & 15, hi8 = (lane & 16) ? 8 : 0;
#pragma unroll
  for (int nt = 0; nt < 4; ++nt) {
    float bias = b1[nt * 16 + n];
    v8f c;
#pragma unroll
    for (int r = 0; r < 8; ++r) c[r] = bias;
#pragma unroll
    for (int kt = 0; kt < 2; ++kt) {
      v16h a = load_a_f16(&sA[w][kt * 32], HID, lane);
      v16h b = load_b_swz(sW1, kt * 4 + nt, lane);
      c = wmma16(a, b, c);
    }
#pragma unroll
    for (int r = 0; r < 8; ++r)
      sM[w][(r + hi8) * HID + nt * 16 + n] = fmaxf(c[r], 0.0f);
  }
  __syncthreads();
#pragma unroll
  for (int nt = 0; nt < 4; ++nt) {
    float bias = b2[nt * 16 + n];
    v8f c;
#pragma unroll
    for (int r = 0; r < 8; ++r) c[r] = bias;
#pragma unroll
    for (int kt = 0; kt < 2; ++kt) {
      v16h a = load_a_f32(&sM[w][kt * 32], HID, lane);
      v16h b = load_b_swz(sW2, kt * 4 + nt, lane);
      c = wmma16(a, b, c);
    }
    float s = 0.f;                      // segment_sum over the 16 edges (rows)
#pragma unroll
    for (int r = 0; r < 8; ++r) s += c[r];
    s += __shfl_xor(s, 16, 32);
    if (lane < 16) msg[node * HID + nt * 16 + lane] = s;
  }
}

// ---------------- Fused flash cross-attention:  mu = Q - softmax(QKᵀ)K ----------------

__global__ __launch_bounds__(128) void attn_kernel(
    const f16* __restrict__ h0h, const f16* __restrict__ h1h,
    const f16* __restrict__ h0T, const f16* __restrict__ h1T,
    const float* __restrict__ h0, const float* __restrict__ h1,
    float* __restrict__ mu0, float* __restrict__ mu1) {
  __shared__ f16 sP[4][16 * 32];
  int dir = blockIdx.z;
  const f16*  Qh  = dir ? h1h : h0h;
  const f16*  Kh  = dir ? h0h : h1h;
  const f16*  KT  = dir ? h0T : h1T;
  const float* Qf = dir ? h1  : h0;
  float*      out = dir ? mu1 : mu0;
  int b = blockIdx.y;
  int w = threadIdx.x >> 5, lane = threadIdx.x & 31;
  int q0 = (blockIdx.x * 4 + w) * 16;               // row within batch sample
  const f16* Qbase = Qh + (size_t)b * NPTS * DIM;
  const f16* Kbase = Kh + (size_t)b * NPTS * DIM;
  const f16* KTb   = KT + (size_t)b * DIM * NPTS;

  v16h aq = load_a_f16(Qbase + q0 * DIM, DIM, lane);
  float mrun[8], lrun[8];
  v8f o0 = {}, o1 = {};
#pragma unroll
  for (int r = 0; r < 8; ++r) { mrun[r] = -1e30f; lrun[r] = 0.f; }
  int n = lane & 15, hi8 = (lane & 16) ? 8 : 0, kh = lane & 16;

  for (int jt = 0; jt < NPTS; jt += 32) {
    v16h bk0 = load_bT_row(Kbase + jt * DIM, DIM, lane);          // S cols jt..jt+15
    v16h bk1 = load_bT_row(Kbase + (jt + 16) * DIM, DIM, lane);   // S cols jt+16..
    v8f z0 = {}, z1 = {};
    v8f s0 = wmma16(aq, bk0, z0);
    v8f s1 = wmma16(aq, bk1, z1);
#pragma unroll
    for (int r = 0; r < 8; ++r) {
      float v = fmaxf(s0[r], s1[r]);                // row-max across 16 lanes
      v = fmaxf(v, __shfl_xor(v, 1, 32));
      v = fmaxf(v, __shfl_xor(v, 2, 32));
      v = fmaxf(v, __shfl_xor(v, 4, 32));
      v = fmaxf(v, __shfl_xor(v, 8, 32));
      float mn = fmaxf(mrun[r], v);
      float sc = __expf(mrun[r] - mn);
      mrun[r] = mn;
      float p0 = __expf(s0[r] - mn);
      float p1 = __expf(s1[r] - mn);
      float rs = p0 + p1;                           // row-sum across 16 lanes
      rs += __shfl_xor(rs, 1, 32);
      rs += __shfl_xor(rs, 2, 32);
      rs += __shfl_xor(rs, 4, 32);
      rs += __shfl_xor(rs, 8, 32);
      lrun[r] = lrun[r] * sc + rs;
      o0[r] *= sc; o1[r] *= sc;
      sP[w][(r + hi8) * 32 + n]      = (f16)p0;     // repack P to A-layout via LDS
      sP[w][(r + hi8) * 32 + 16 + n] = (f16)p1;
    }
    v16h ap  = load_a_f16(sP[w], 32, lane);
    // O accumulation: B[k][d] = K[jt+k][d] = KT[d][jt+k] -> contiguous in k
    v16h bo0 = *(const v16h*)(KTb + (size_t)n * NPTS + jt + kh);          // d 0..15
    v16h bo1 = *(const v16h*)(KTb + (size_t)(16 + n) * NPTS + jt + kh);   // d 16..31
    o0 = wmma16(ap, bo0, o0);
    o1 = wmma16(ap, bo1, o1);
  }
#pragma unroll
  for (int r = 0; r < 8; ++r) {
    float inv = 1.0f / lrun[r];
    int row = b * NPTS + q0 + r + hi8;
    out[row * DIM + n]      = Qf[row * DIM + n]      - o0[r] * inv;
    out[row * DIM + 16 + n] = Qf[row * DIM + 16 + n] - o1[r] * inv;
  }
}

// ---------------- Node MLP: [h||msg||mu] (128) -> 64 relu -> 32 ----------------

__global__ __launch_bounds__(256) void node_mlp_kernel(
    const float* __restrict__ h, const float* __restrict__ msgb,
    const float* __restrict__ mub,
    const f16* __restrict__ W1s, const float* __restrict__ b1,
    const f16* __restrict__ W2s, const float* __restrict__ b2,
    float* __restrict__ hout) {
  __shared__ f16  sW1[128 * HID];       // swizzled 128x64
  __shared__ f16  sW2[HID * DIM];       // swizzled 64x32
  __shared__ f16  sA[8][16 * 128];
  __shared__ float sM[8][16 * HID];
  int tid = threadIdx.x;
  for (int i = tid; i < 128 * HID / 8; i += 256) ((v8h*)sW1)[i] = ((const v8h*)W1s)[i];
  for (int i = tid; i < HID * DIM / 8; i += 256) ((v8h*)sW2)[i] = ((const v8h*)W2s)[i];
  int w = tid >> 5, lane = tid & 31;
  int n0 = (blockIdx.x * 8 + w) * 16;   // 16 nodes per wave
  // stage A: 16 rows x 16 chunks of 8 (concat regions are 8-aligned)
  for (int idx = lane; idx < 256; idx += 32) {
    int r = idx >> 4, cc = (idx & 15) << 3;
    int node = n0 + r;
    const float* srcp = (cc < 32) ? (h + (size_t)node * DIM + cc)
                      : (cc < 96) ? (msgb + (size_t)node * HID + (cc - 32))
                                  : (mub + (size_t)node * DIM + (cc - 96));
    *(v8h*)(&sA[w][r * 128 + cc]) = __builtin_convertvector(*(const v8f*)srcp, v8h);
  }
  __syncthreads();
  int n = lane & 15, hi8 = (lane & 16) ? 8 : 0;
#pragma unroll
  for (int nt = 0; nt < 4; ++nt) {
    float bias = b1[nt * 16 + n];
    v8f c;
#pragma unroll
    for (int r = 0; r < 8; ++r) c[r] = bias;
#pragma unroll
    for (int kt = 0; kt < 4; ++kt) {
      v16h a = load_a_f16(&sA[w][kt * 32], 128, lane);
      v16h b = load_b_swz(sW1, kt * 4 + nt, lane);
      c = wmma16(a, b, c);
    }
#pragma unroll
    for (int r = 0; r < 8; ++r)
      sM[w][(r + hi8) * HID + nt * 16 + n] = fmaxf(c[r], 0.0f);
  }
  __syncthreads();
#pragma unroll
  for (int nt = 0; nt < 2; ++nt) {
    float bias = b2[nt * 16 + n];
    v8f c;
#pragma unroll
    for (int r = 0; r < 8; ++r) c[r] = bias;
#pragma unroll
    for (int kt = 0; kt < 2; ++kt) {
      v16h a = load_a_f32(&sM[w][kt * 32], HID, lane);
      v16h b = load_b_swz(sW2, kt * 2 + nt, lane);
      c = wmma16(a, b, c);
    }
#pragma unroll
    for (int r = 0; r < 8; ++r)
      hout[(n0 + r + hi8) * DIM + nt * 16 + n] = c[r];
  }
}

// ---------------- Host orchestration ----------------

extern "C" void kernel_launch(void* const* d_in, const int* in_sizes, int n_in,
                              void* d_out, int out_size, void* d_ws, size_t ws_size,
                              hipStream_t stream) {
  (void)in_sizes; (void)n_in; (void)out_size; (void)ws_size;
  const float* input_xyz = (const float*)d_in[0];
  const float* coord_xyz = (const float*)d_in[1];
  const float* input_f   = (const float*)d_in[2];
  const float* coord_f   = (const float*)d_in[3];
  const float* Wf = (const float*)d_in[4];
  const float* bf = (const float*)d_in[5];
  const float* Ws = (const float*)d_in[6];
  const float* bs = (const float*)d_in[7];
  const float* gamma = (const float*)d_in[8];
  const float* beta  = (const float*)d_in[9];
  const float* ew1 = (const float*)d_in[10];
  const float* eb1 = (const float*)d_in[11];
  const float* ew2 = (const float*)d_in[12];
  const float* eb2 = (const float*)d_in[13];
  const float* nw1 = (const float*)d_in[14];
  const float* nb1 = (const float*)d_in[15];
  const float* nw2 = (const float*)d_in[16];
  const float* nb2 = (const float*)d_in[17];

  size_t off = 0;
  char* base = (char*)d_ws;
  auto alloc = [&](size_t bytes) -> void* {
    void* p = base + off;
    off += (bytes + 255) & ~(size_t)255;
    return p;
  };
  int*   nbr0 = (int*)alloc((size_t)NN * KNN * 4);
  int*   nbr1 = (int*)alloc((size_t)NN * KNN * 4);
  float* hA0  = (float*)alloc((size_t)NN * DIM * 4);
  float* hA1  = (float*)alloc((size_t)NN * DIM * 4);
  float* hB0  = (float*)alloc((size_t)NN * DIM * 4);
  float* hB1  = (float*)alloc((size_t)NN * DIM * 4);
  float* agg  = (float*)alloc((size_t)NN * DIM * 4);
  float* stats = (float*)alloc(64 * 4);
  float* msg0 = (float*)alloc((size_t)NN * HID * 4);
  float* msg1 = (float*)alloc((size_t)NN * HID * 4);
  float* mu0  = (float*)alloc((size_t)NN * DIM * 4);
  float* mu1  = (float*)alloc((size_t)NN * DIM * 4);
  f16*   h0h  = (f16*)alloc((size_t)NN * DIM * 2);
  f16*   h1h  = (f16*)alloc((size_t)NN * DIM * 2);
  f16*   h0T  = (f16*)alloc((size_t)NN * DIM * 2);
  f16*   h1T  = (f16*)alloc((size_t)NN * DIM * 2);
  f16*   ew1s = (f16*)alloc((size_t)5 * HID * HID * 2);
  f16*   ew2s = (f16*)alloc((size_t)5 * HID * HID * 2);
  f16*   nw1s = (f16*)alloc((size_t)5 * 128 * HID * 2);
  f16*   nw2s = (f16*)alloc((size_t)5 * HID * DIM * 2);

  // 1. kNN graphs
  knn_kernel<<<NN, 256, 0, stream>>>(input_xyz, nbr0);
  knn_kernel<<<NN, 256, 0, stream>>>(coord_xyz, nbr1);

  // 2. CGConv on both graphs (shared conv1 weights)
  cgconv_kernel<<<(NN * 32) / 256, 256, 0, stream>>>(input_f, nbr0, Wf, bf, Ws, bs, agg);
  bn_stats_kernel<<<1, 256, 0, stream>>>(agg, stats);
  bn_apply_kernel<<<(NN * DIM) / 256, 256, 0, stream>>>(input_f, agg, stats, gamma, beta, hA0);
  cgconv_kernel<<<(NN * 32) / 256, 256, 0, stream>>>(coord_f, nbr1, Wf, bf, Ws, bs, agg);
  bn_stats_kernel<<<1, 256, 0, stream>>>(agg, stats);
  bn_apply_kernel<<<(NN * DIM) / 256, 256, 0, stream>>>(coord_f, agg, stats, gamma, beta, hA1);

  // 3. Pre-swizzle MLP weights for fragment-major LDS staging (all 5 layers)
  swz_b_kernel<<<(5 * HID * HID + 255) / 256, 256, 0, stream>>>(ew1, ew1s, HID, HID, 5 * HID * HID);
  swz_b_kernel<<<(5 * HID * HID + 255) / 256, 256, 0, stream>>>(ew2, ew2s, HID, HID, 5 * HID * HID);
  swz_b_kernel<<<(5 * 128 * HID + 255) / 256, 256, 0, stream>>>(nw1, nw1s, 128, HID, 5 * 128 * HID);
  swz_b_kernel<<<(5 * HID * DIM + 255) / 256, 256, 0, stream>>>(nw2, nw2s, HID, DIM, 5 * HID * DIM);

  // 4. Five propagation layers
  float* cur0 = hA0; float* cur1 = hA1;
  float* nxt0 = hB0; float* nxt1 = hB1;
  for (int l = 0; l < 5; ++l) {
    tof16_tr_kernel<<<(NN * DIM) / 256, 256, 0, stream>>>(cur0, h0h, h0T);
    tof16_tr_kernel<<<(NN * DIM) / 256, 256, 0, stream>>>(cur1, h1h, h1T);

    edge_mlp_kernel<<<NN / 8, 256, 0, stream>>>(
        h0h, nbr0, ew1s + (size_t)l * HID * HID, eb1 + l * HID,
        ew2s + (size_t)l * HID * HID, eb2 + l * HID, msg0);
    edge_mlp_kernel<<<NN / 8, 256, 0, stream>>>(
        h1h, nbr1, ew1s + (size_t)l * HID * HID, eb1 + l * HID,
        ew2s + (size_t)l * HID * HID, eb2 + l * HID, msg1);

    attn_kernel<<<dim3(NPTS / 64, NB, 2), 128, 0, stream>>>(h0h, h1h, h0T, h1T,
                                                            cur0, cur1, mu0, mu1);

    node_mlp_kernel<<<NN / 128, 256, 0, stream>>>(
        cur0, msg0, mu0, nw1s + (size_t)l * 128 * HID, nb1 + l * HID,
        nw2s + (size_t)l * HID * DIM, nb2 + l * DIM, nxt0);
    node_mlp_kernel<<<NN / 128, 256, 0, stream>>>(
        cur1, msg1, mu1, nw1s + (size_t)l * 128 * HID, nb1 + l * HID,
        nw2s + (size_t)l * HID * DIM, nb2 + l * DIM, nxt1);

    float* t0 = cur0; cur0 = nxt0; nxt0 = t0;
    float* t1 = cur1; cur1 = nxt1; nxt1 = t1;
  }

  // 5. Emit (f0, f1) concatenated
  hipMemcpyAsync(d_out, cur0, (size_t)NN * DIM * 4, hipMemcpyDeviceToDevice, stream);
  hipMemcpyAsync((float*)d_out + (size_t)NN * DIM, cur1, (size_t)NN * DIM * 4,
                 hipMemcpyDeviceToDevice, stream);
}
